// LSTM_395136991665
// MI455X (gfx1250) — matrix operand order repaired and checked by reading, not compiled, not run on hardware
//
#include <hip/hip_runtime.h>
#include <hip/hip_bf16.h>
#include <stdint.h>

// ---------------------------------------------------------------------------
// Types for CDNA5 WMMA (wave32): D(f32 16x16) = A(bf16 16x32) x B(bf16 32x16) + C
// ---------------------------------------------------------------------------
typedef __attribute__((ext_vector_type(16))) __bf16 v16bf;
typedef __attribute__((ext_vector_type(8)))  float  v8f;

union FragU {
    uint4 q[2];   // two 16B chunks loaded from LDS
    v16bf v;      // viewed as the WMMA operand
};

#if __has_builtin(__builtin_amdgcn_global_load_async_to_lds_b128)
#define HAVE_ASYNC_LDS 1
#else
#define HAVE_ASYNC_LDS 0
#endif

// Pointer types matching the async-to-LDS builtin signature:
//   void __builtin_amdgcn_global_load_async_to_lds_b128(v4i AS1*, v4i AS3*, Imm, Imm)
typedef int v4i_ __attribute__((vector_size(16)));
typedef __attribute__((address_space(1))) v4i_ g_v4i;
typedef __attribute__((address_space(3))) v4i_ l_v4i;

// 16-byte global -> LDS copy (async DMA on CDNA5 when available)
__device__ __forceinline__ void cp16_g2l(const void* g, unsigned short* lds) {
#if HAVE_ASYNC_LDS
    __builtin_amdgcn_global_load_async_to_lds_b128(
        (g_v4i*)(g), (l_v4i*)(lds), 0, 0);
#else
    *(uint4*)lds = *(const uint4*)g;
#endif
}

__device__ __forceinline__ void async_wait0() {
#if HAVE_ASYNC_LDS
#if __has_builtin(__builtin_amdgcn_s_wait_asynccnt)
    __builtin_amdgcn_s_wait_asynccnt(0);
#else
    asm volatile("s_wait_asynccnt 0" ::: "memory");
#endif
#endif
}

__device__ __forceinline__ unsigned short f2bf(float f) {
    union { float f; uint32_t u; } v; v.f = f;
    uint32_t u = v.u;
    return (unsigned short)((u + 0x7FFFu + ((u >> 16) & 1u)) >> 16);  // RNE
}
__device__ __forceinline__ float sigmoidf_(float x) {
    return 1.0f / (1.0f + __expf(-x));
}
__device__ __forceinline__ float fast_tanh(float x) {
#if __has_builtin(__builtin_amdgcn_tanhf)
    return __builtin_amdgcn_tanhf(x);          // v_tanh_f32 (gfx1250)
#elif __has_builtin(__builtin_amdgcn_tanh_f32)
    return __builtin_amdgcn_tanh_f32(x);
#else
    float t = __expf(-2.0f * fabsf(x));
    float r = (1.0f - t) / (1.0f + t);
    return copysignf(r, x);
#endif
}

// ---------------------------------------------------------------------------
// Fused LSTM step:  G = A0 @ W0^T + A1 @ W1^T + bias  (two K-phases, same acc)
//                   c' = sig(Gf)*c + sig(Gi)*tanh(Gg);  h = sig(Go)*tanh(c')
// Block tile: 64 rows x 64 cols x 4 gates.  Grid: (B/64, 512/64).
// 8 waves: wave = gate(0..3) x m-half(0..1); per wave 2x4 WMMA accumulators.
// ---------------------------------------------------------------------------
__launch_bounds__(256)
__global__ void lstm_step_kernel(
    const void* __restrict__ A0p, int a0f32, int lda0, int K0,
    const unsigned short* __restrict__ W0, int ldw0,
    const void* __restrict__ A1p, int a1f32, int lda1, int K1,
    const unsigned short* __restrict__ W1, int ldw1,
    const float* __restrict__ bias,
    float* __restrict__ c,
    unsigned short* __restrict__ h_bf16,
    float* __restrict__ h_f32)
{
    __shared__ union SMem {
        struct { unsigned short A[64 * 72]; unsigned short W[4 * 64 * 72]; } s;
        float G[32 * 260];   // half-tile gate exchange buffer
    } sm;

    const int tid  = threadIdx.x;
    const int m0   = blockIdx.x * 64;
    const int n0   = blockIdx.y * 64;
    const int wave = tid >> 5;
    const int lane = tid & 31;
    const int hl   = lane >> 4;      // lane half (0/1)
    const int r    = lane & 15;
    const int gate = wave & 3;
    const int mh   = wave >> 2;      // 0/1: which 32 rows of the 64-row tile

    v8f acc[2][4];
#pragma unroll
    for (int i = 0; i < 2; ++i)
#pragma unroll
        for (int j = 0; j < 4; ++j) acc[i][j] = {};

    for (int phase = 0; phase < 2; ++phase) {
        const void*           Ap    = phase ? A1p   : A0p;
        const unsigned short* Wp    = phase ? W1    : W0;
        const int             isf32 = phase ? a1f32 : a0f32;
        const int             lda   = phase ? lda1  : lda0;
        const int             ldw   = phase ? ldw1  : ldw0;
        const int             K     = phase ? K1    : K0;
        if (K == 0) continue;

        for (int k0 = 0; k0 < K; k0 += 64) {
            // ---- stage A tile (64 rows x 64 k) into LDS, converting if f32 ----
            {
                const int row = tid >> 2;            // 0..63
                const int cg  = (tid & 3) * 16;      // 16 k-elements per thread
                unsigned short* dst = &sm.s.A[row * 72 + cg];
                if (isf32) {
                    const float* src = (const float*)Ap + (size_t)(m0 + row) * lda + k0 + cg;
#pragma unroll
                    for (int j = 0; j < 4; ++j) {
                        float4 f = *(const float4*)(src + j * 4);
                        ushort4 h;
                        h.x = f2bf(f.x); h.y = f2bf(f.y);
                        h.z = f2bf(f.z); h.w = f2bf(f.w);
                        *(ushort4*)(dst + j * 4) = h;
                    }
                } else {
                    const unsigned short* src =
                        (const unsigned short*)Ap + (size_t)(m0 + row) * lda + k0 + cg;
                    cp16_g2l(src,     dst);
                    cp16_g2l(src + 8, dst + 8);
                }
            }
            // ---- stage W tiles: 4 gates x 64 n-rows x 64 k ----
            {
                const int g    = tid >> 6;           // 0..3
                const int nrow = tid & 63;
                const unsigned short* src =
                    Wp + (size_t)(g * 512 + n0 + nrow) * ldw + k0;
                unsigned short* dst = &sm.s.W[(g * 64 + nrow) * 72];
#pragma unroll
                for (int j = 0; j < 8; ++j)
                    cp16_g2l(src + j * 8, dst + j * 8);
            }
            async_wait0();
            __syncthreads();

            // ---- WMMA compute: 2 k-steps of 32, 2x4 tiles per wave ----
#pragma unroll
            for (int ks = 0; ks < 64; ks += 32) {
                FragU a[2], b[4];
                // A operand lane layout (16-bit A 16x32, ISA 7.12.2):
                //   elements 0..7  -> K = ks + hl*8  + e
                //   elements 8..15 -> K = ks + 16 + hl*8 + (e-8)
#pragma unroll
                for (int mt = 0; mt < 2; ++mt) {
                    const unsigned short* ap =
                        &sm.s.A[(mh * 32 + mt * 16 + r) * 72 + ks + hl * 8];
                    a[mt].q[0] = *(const uint4*)(ap);
                    a[mt].q[1] = *(const uint4*)(ap + 16);
                }
                // B operand lane layout (same as A, lane%16 = column):
                //   elements 0..7 -> K = ks + hl*8 + e; elements 8..15 -> K = ks + 16 + hl*8 + (e-8)
#pragma unroll
                for (int nt = 0; nt < 4; ++nt) {
                    const unsigned short* bp =
                        &sm.s.W[(gate * 64 + nt * 16 + r) * 72 + ks + hl * 8];
                    b[nt].q[0] = *(const uint4*)(bp);
                    b[nt].q[1] = *(const uint4*)(bp + 16);
                }
#pragma unroll
                for (int mt = 0; mt < 2; ++mt)
#pragma unroll
                    for (int nt = 0; nt < 4; ++nt)
                        acc[mt][nt] = __builtin_amdgcn_wmma_f32_16x16x32_bf16(
                            false, a[mt].v, false, b[nt].v,
                            (short)0, acc[mt][nt], false, false);
            }
            __syncthreads();
        }
    }

    // ---- fused gate epilogue, one 32-row half at a time (LDS exchange) ----
    for (int mhp = 0; mhp < 2; ++mhp) {
        if (mh == mhp) {
#pragma unroll
            for (int nt = 0; nt < 4; ++nt) {
                const float bv = bias[gate * 512 + n0 + nt * 16 + r];
#pragma unroll
                for (int mt = 0; mt < 2; ++mt)
#pragma unroll
                    for (int v = 0; v < 8; ++v) {
                        const int ml = mt * 16 + hl * 8 + v;   // 0..31
                        sm.G[ml * 260 + gate * 64 + nt * 16 + r] =
                            acc[mt][nt][v] + bv;
                    }
            }
        }
        __syncthreads();
        {
            const int n     = tid & 63;
            const int mbase = (tid >> 6) * 8;
#pragma unroll
            for (int j = 0; j < 8; ++j) {
                const int ml = mbase + j;
                const int m  = mhp * 32 + ml;
                const float gi = sm.G[ml * 260 +       n];
                const float gf = sm.G[ml * 260 +  64 + n];
                const float gg = sm.G[ml * 260 + 128 + n];
                const float go = sm.G[ml * 260 + 192 + n];
                const size_t idx = (size_t)(m0 + m) * 512 + (n0 + n);
                const float cv = sigmoidf_(gf) * c[idx] + sigmoidf_(gi) * fast_tanh(gg);
                const float hv = sigmoidf_(go) * fast_tanh(cv);
                c[idx]      = cv;
                h_bf16[idx] = f2bf(hv);
                if (h_f32) h_f32[idx] = hv;
            }
        }
        __syncthreads();
    }
}

// ---------------------------------------------------------------------------
// Small helper kernels
// ---------------------------------------------------------------------------
__global__ void cvt_f32_bf16(const float* __restrict__ src,
                             unsigned short* __restrict__ dst, long long n) {
    long long i = (long long)blockIdx.x * blockDim.x + threadIdx.x;
    const long long stride = (long long)gridDim.x * blockDim.x;
    for (; i < n; i += stride) dst[i] = f2bf(src[i]);
}

__global__ void add_vec(const float* __restrict__ a, const float* __restrict__ b,
                        float* __restrict__ o, int n) {
    int i = blockIdx.x * blockDim.x + threadIdx.x;
    if (i < n) o[i] = a[i] + b[i];
}

__global__ void zero_u32(uint32_t* __restrict__ p, long long n) {
    long long i = (long long)blockIdx.x * blockDim.x + threadIdx.x;
    const long long stride = (long long)gridDim.x * blockDim.x;
    for (; i < n; i += stride) p[i] = 0u;
}

// Output head: obj = h2 @ W + b  (nout = 35 or 82).  If emb != null, also does
// argmax -> embedding gather and builds xin = [outs1_t | emb[arg]] in bf16.
__launch_bounds__(128)
__global__ void head_kernel(const float* __restrict__ h2,
                            const float* __restrict__ W,
                            const float* __restrict__ b, int nout,
                            float* __restrict__ out, int out_off,
                            const float* __restrict__ emb,
                            const unsigned short* __restrict__ outs1_t,
                            unsigned short* __restrict__ xin)
{
    __shared__ float hrow[512];
    __shared__ float logits[96];
    __shared__ int   argi;
    const int row = blockIdx.x;
    const int tid = threadIdx.x;

#pragma unroll
    for (int j = 0; j < 4; ++j)
        hrow[tid * 4 + j] = h2[(size_t)row * 512 + tid * 4 + j];
    __syncthreads();

    if (tid < nout) {
        float s = b[tid];
        for (int k = 0; k < 512; ++k) s += hrow[k] * W[k * nout + tid];
        out[(size_t)row * 152 + out_off + tid] = s;
        logits[tid] = s;
    }
    __syncthreads();

    if (emb != nullptr) {
        if (tid == 0) {
            int best = 0; float bv = logits[0];
            for (int k = 1; k < nout; ++k)
                if (logits[k] > bv) { bv = logits[k]; best = k; }
            argi = best;
        }
        __syncthreads();
        if (tid < 64)
            xin[(size_t)row * 576 + 512 + tid] = f2bf(emb[argi * 64 + tid]);
#pragma unroll
        for (int j = 0; j < 4; ++j)
            xin[(size_t)row * 576 + tid * 4 + j] =
                outs1_t[(size_t)row * 512 + tid * 4 + j];
    }
}

// ---------------------------------------------------------------------------
// Host-side orchestration
// ---------------------------------------------------------------------------
extern "C" void kernel_launch(void* const* d_in, const int* in_sizes, int n_in,
                              void* d_out, int out_size, void* d_ws, size_t ws_size,
                              hipStream_t stream)
{
    (void)in_sizes; (void)n_in; (void)out_size; (void)ws_size;

    const float* x    = (const float*)d_in[0];    // [33,4096,512]
    const float* Wih1 = (const float*)d_in[1];    // [2048,512]
    const float* Whh1 = (const float*)d_in[2];    // [2048,512]
    const float* bih1 = (const float*)d_in[3];    // [2048]
    const float* bhh1 = (const float*)d_in[4];    // [2048]
    const float* Wih2 = (const float*)d_in[5];    // [33,2048,576]
    const float* Whh2 = (const float*)d_in[6];    // [33,2048,512]
    const float* bih2 = (const float*)d_in[7];    // [33,2048]
    const float* bhh2 = (const float*)d_in[8];    // [33,2048]
    const float* Wo1  = (const float*)d_in[9];    // [512,35]
    const float* bo1  = (const float*)d_in[10];
    const float* Wo2  = (const float*)d_in[11];
    const float* bo2  = (const float*)d_in[12];
    const float* Wr   = (const float*)d_in[13];   // [512,82]
    const float* br   = (const float*)d_in[14];
    const float* emb  = (const float*)d_in[15];   // [35,64]
    float* out = (float*)d_out;                   // [4096,152]

    // ---- workspace carve-up (256B aligned) ----
    char* ws = (char*)d_ws;
    auto carve = [&](size_t bytes) -> char* {
        char* p = ws; ws += (bytes + 255) & ~(size_t)255; return p;
    };
    unsigned short* wih1  = (unsigned short*)carve(2048ull * 512 * 2);
    unsigned short* whh1  = (unsigned short*)carve(2048ull * 512 * 2);
    unsigned short* wih2  = (unsigned short*)carve(33ull * 2048 * 576 * 2);
    unsigned short* whh2  = (unsigned short*)carve(33ull * 2048 * 512 * 2);
    float*          bias1 = (float*)carve(2048ull * 4);
    float*          bias2 = (float*)carve(33ull * 2048 * 4);
    unsigned short* outs1 = (unsigned short*)carve(33ull * 4096 * 512 * 2);
    unsigned short* zbuf  = (unsigned short*)carve(4096ull * 512 * 2);
    unsigned short* h2a   = (unsigned short*)carve(4096ull * 512 * 2);
    unsigned short* h2b   = (unsigned short*)carve(4096ull * 512 * 2);
    float*          h2f   = (float*)carve(4096ull * 512 * 4);
    float*          c1    = (float*)carve(4096ull * 512 * 4);
    float*          c2    = (float*)carve(4096ull * 512 * 4);
    unsigned short* xin   = (unsigned short*)carve(4096ull * 576 * 2);

    // ---- precompute: bf16 weights, combined biases, zero state ----
    cvt_f32_bf16<<<2048, 256, 0, stream>>>(Wih1, wih1, 2048LL * 512);
    cvt_f32_bf16<<<2048, 256, 0, stream>>>(Whh1, whh1, 2048LL * 512);
    cvt_f32_bf16<<<4096, 256, 0, stream>>>(Wih2, wih2, 33LL * 2048 * 576);
    cvt_f32_bf16<<<4096, 256, 0, stream>>>(Whh2, whh2, 33LL * 2048 * 512);
    add_vec<<<8, 256, 0, stream>>>(bih1, bhh1, bias1, 2048);
    add_vec<<<264, 256, 0, stream>>>(bih2, bhh2, bias2, 33 * 2048);
    zero_u32<<<1024, 256, 0, stream>>>((uint32_t*)zbuf, 4096LL * 512 / 2);
    zero_u32<<<1024, 256, 0, stream>>>((uint32_t*)c1, 4096LL * 512);
    zero_u32<<<1024, 256, 0, stream>>>((uint32_t*)c2, 4096LL * 512);

    const dim3 sgrid(4096 / 64, 512 / 64);  // (64, 8)

    // ---- stage 1: standard LSTM, 33 steps ----
    for (int t = 0; t < 33; ++t) {
        const void* Aprev = (t == 0) ? (const void*)zbuf
                                     : (const void*)(outs1 + (size_t)(t - 1) * 4096 * 512);
        lstm_step_kernel<<<sgrid, 256, 0, stream>>>(
            (const void*)(x + (size_t)t * 4096 * 512), 1, 512, 512, wih1, 512,
            Aprev, 0, 512, 512, whh1, 512,
            bias1, c1, outs1 + (size_t)t * 4096 * 512, nullptr);
    }

    // ---- stage 2: 31 distinct LSTMCells with zero word-vec ----
    const unsigned short* prev = zbuf;
    for (int t = 0; t < 31; ++t) {
        unsigned short* cur = (t & 1) ? h2b : h2a;
        lstm_step_kernel<<<sgrid, 256, 0, stream>>>(
            (const void*)(outs1 + (size_t)t * 4096 * 512), 0, 512, 512,
            wih2 + (size_t)t * 2048 * 576, 576,
            (const void*)prev, 0, 512, 512,
            whh2 + (size_t)t * 2048 * 512, 512,
            bias2 + (size_t)t * 2048, c2, cur, h2f);
        prev = cur;
    }

    // ---- obj1 head + argmax feedback + step 31 (K0 = 576 full concat) ----
    head_kernel<<<4096, 128, 0, stream>>>(h2f, Wo1, bo1, 35, out, 0,
                                          emb, outs1 + 31ull * 4096 * 512, xin);
    {
        unsigned short* cur = h2b;  // t=31 (odd)
        lstm_step_kernel<<<sgrid, 256, 0, stream>>>(
            (const void*)xin, 0, 576, 576, wih2 + 31ull * 2048 * 576, 576,
            (const void*)prev, 0, 512, 512, whh2 + 31ull * 2048 * 512, 512,
            bias2 + 31ull * 2048, c2, cur, h2f);
        prev = cur;
    }

    // ---- obj2 head + argmax feedback + step 32 ----
    head_kernel<<<4096, 128, 0, stream>>>(h2f, Wo2, bo2, 35, out, 35,
                                          emb, outs1 + 32ull * 4096 * 512, xin);
    {
        unsigned short* cur = h2a;  // t=32 (even)
        lstm_step_kernel<<<sgrid, 256, 0, stream>>>(
            (const void*)xin, 0, 576, 576, wih2 + 32ull * 2048 * 576, 576,
            (const void*)prev, 0, 512, 512, whh2 + 32ull * 2048 * 512, 512,
            bias2 + 32ull * 2048, c2, cur, h2f);
        prev = cur;
    }

    // ---- relation head ----
    head_kernel<<<4096, 128, 0, stream>>>(h2f, Wr, br, 82, out, 70,
                                          nullptr, nullptr, nullptr);
}
